// NonLocalBlock_57062935494904
// MI455X (gfx1250) — matrix operand order repaired and checked by reading, not compiled, not run on hardware
//
#include <hip/hip_runtime.h>

#define DEVI __device__ __forceinline__

typedef __attribute__((ext_vector_type(16))) _Float16 v16h;
typedef __attribute__((ext_vector_type(8)))  _Float16 v8h;
typedef __attribute__((ext_vector_type(8)))  float    v8f;
typedef __attribute__((ext_vector_type(4)))  unsigned int v4u;
typedef __attribute__((ext_vector_type(8)))  int      v8i;
typedef __attribute__((ext_vector_type(4)))  int      v4i;

static constexpr int Bb = 8;     // batch
static constexpr int C  = 256;   // channels
static constexpr int Ci = 128;   // inner channels
static constexpr int N  = 4096;  // H*W

DEVI v8f vzero() {
  v8f v = {0.f,0.f,0.f,0.f,0.f,0.f,0.f,0.f};
  return v;
}

// ---- WMMA operand fragment loaders (CDNA5 16-bit layouts, ISA 7.12.2) ----
// A-matrix 16x32 f16: row = lane%16; lanes<16 hold K={k0..k0+7, k0+16..k0+23},
// lanes>=16 hold K shifted by +8. Two 16B loads.
DEVI v16h frag_A(const _Float16* m, int ld, int k0) {
  int l = threadIdx.x & 31;
  const _Float16* p = m + (size_t)(l & 15) * ld + k0 + ((l >> 4) << 3);
  union { v16h v; v8h h[2]; } u;
  u.h[0] = *(const v8h*)(p);
  u.h[1] = *(const v8h*)(p + 16);
  return u.v;
}

// B-matrix 32x16 f16 stored transposed: element B[k][c] at m[c*ld + k].
// column c = lane%16; lanes<16 hold K={k0..k0+7, k0+16..k0+23}, lanes>=16 K shifted by +8.
DEVI v16h frag_B(const _Float16* m, int ld, int k0) {
  int l = threadIdx.x & 31;
  const _Float16* p = m + (size_t)(l & 15) * ld + k0 + ((l >> 4) << 3);
  union { v16h v; v8h h[2]; } u;
  u.h[0] = *(const v8h*)(p);
  u.h[1] = *(const v8h*)(p + 16);
  return u.v;
}

DEVI v8f wmma(v16h a, v16h b, v8f c) {
  // (neg_a, A, neg_b, B, c_mod, C, reuse_a, reuse_b)
  return __builtin_amdgcn_wmma_f32_16x16x32_f16(false, a, false, b, (short)0, c, false, false);
}

// ---- Tensor Data Mover helpers (ISA ch.7/8: TENSOR_LOAD_TO_LDS + D#) ----
#if __has_builtin(__builtin_amdgcn_tensor_load_to_lds)
#define HAVE_TDM 1
#else
#define HAVE_TDM 0
#endif

DEVI void wait_tensorcnt0() {
#if __has_builtin(__builtin_amdgcn_s_wait_tensorcnt)
  __builtin_amdgcn_s_wait_tensorcnt(0);
#else
  asm volatile("s_wait_tensorcnt 0x0" ::: "memory");
#endif
}
DEVI void wait_tensorcnt2() {
#if __has_builtin(__builtin_amdgcn_s_wait_tensorcnt)
  __builtin_amdgcn_s_wait_tensorcnt(2);
#else
  asm volatile("s_wait_tensorcnt 0x2" ::: "memory");
#endif
}

// LDS byte offset: generic LDS pointer = {SHARED_BASE[63:32], offset[31:0]}
DEVI unsigned lds_byte_addr(const void* p) {
  return (unsigned)(size_t)p;
}

#if HAVE_TDM
// Issue a 2D TDM tile load. grp1 packs (ISA 8.4): data_size=2B,
// tensor_dim0/1, tile_dim0/1, tensor_dim0_stride; no pad/iterate/multicast.
// 6-arg builtin form (clang-23 / therock headers): (g0, g1, g2, g3, g4, cpol)
DEVI void tdm_load_2d(unsigned lds_byte, const void* gaddr, v8i grp1) {
  unsigned long long ga = (unsigned long long)(size_t)gaddr;
  v4u g0;
  g0.x = 1u;                                  // count=1, user mode
  g0.y = lds_byte;                            // lds_addr
  g0.z = (unsigned)(ga & 0xFFFFFFFFull);      // global_addr[31:0]
  g0.w = (unsigned)((ga >> 32) & 0x01FFFFFFull) | 0x80000000u;  // addr[56:32] | type=2
  v4i z4 = {0, 0, 0, 0};                      // groups 2/3 unused (<=2D)
  v8i z8 = {0, 0, 0, 0, 0, 0, 0, 0};
  __builtin_amdgcn_tensor_load_to_lds(g0, grp1, z4, z4, z8, 0);
}

// phi tile: tensor [N rows x Ci], tile [32 x Ci], elem=2B
DEVI v8i grp1_phi() {
  // w0: data_size=1(2B)<<16 ; w1..: dim0=128, dim1=4096, tile0=128, tile1=32, stride0=128
  v8i g = { 0x00010000, 0x00800000, 0x10000000, 0x00800000, 32, 128, 0, 0 };
  return g;
}
// g tile: tensor [Ci rows x N], tile [128 x 32], elem=2B
DEVI v8i grp1_g() {
  // dim0=4096, dim1=128, tile0=32, tile1=128, stride0=4096
  v8i g = { 0x00010000, 0x10000000, 0x00800000, 0x00200000, 128, 4096, 0, 0 };
  return g;
}
#endif

// ---------------- prep: zero BN stats ----------------
__global__ void k_zero(float* p) {
  int i = threadIdx.x;
  if (i < 512) p[i] = 0.f;
}

// ---------------- prep: weights to f16 ----------------
__global__ void k_prep_w(const float* __restrict__ tw, const float* __restrict__ tb,
                         const float* __restrict__ pw, const float* __restrict__ pb,
                         const float* __restrict__ gw, const float* __restrict__ gb,
                         const float* __restrict__ ow,
                         _Float16* __restrict__ Wqkv, _Float16* __restrict__ Wout,
                         float* __restrict__ bqkv) {
  int i = blockIdx.x * blockDim.x + threadIdx.x;
  const int nqkv = 384 * C;
  const int nout = C * Ci;
  if (i < nqkv) {
    int m = i >> 8, k = i & 255;
    float v = (m < 128) ? tw[m * C + k] : (m < 256) ? pw[(m - 128) * C + k] : gw[(m - 256) * C + k];
    Wqkv[i] = (_Float16)v;
  } else if (i < nqkv + nout) {
    Wout[i - nqkv] = (_Float16)ow[i - nqkv];
  } else if (i < nqkv + nout + 384) {
    int m = i - nqkv - nout;
    bqkv[m] = (m < 128) ? tb[m] : (m < 256) ? pb[m - 128] : gb[m - 256];
  }
}

// ---------------- prep: x [B,C,N] fp32 -> xh [B,N,C] f16 (LDS transpose) ----------------
__global__ __launch_bounds__(256) void k_prep_x(const float* __restrict__ x,
                                                _Float16* __restrict__ xh) {
  __shared__ float tile[32][65];
  int id = blockIdx.x;
  int b = id >> 9;
  int rem = id & 511;
  int c0 = (rem >> 6) << 5;   // channel tile of 32
  int n0 = (rem & 63) << 6;   // pixel tile of 64
  int t = threadIdx.x;
#pragma unroll
  for (int rep = 0; rep < 8; ++rep) {
    int idx = rep * 256 + t;
    int r = idx >> 6, cc = idx & 63;
    tile[r][cc] = x[((size_t)(b * C + c0 + r)) * N + n0 + cc];
  }
  __syncthreads();
#pragma unroll
  for (int rep = 0; rep < 8; ++rep) {
    int idx = rep * 256 + t;
    int p = idx >> 5, ch = idx & 31;
    xh[((size_t)(b * N) + n0 + p) * C + c0 + ch] = (_Float16)tile[ch][p];
  }
}

// ---------------- QKV projection: Wqkv[384,256] @ xh^T -> thT/phT [B,N,Ci], gC [B,Ci,N] ----------------
__global__ __launch_bounds__(256) void k_proj(const _Float16* __restrict__ xh,
                                              const _Float16* __restrict__ Wqkv,
                                              const float* __restrict__ bqkv,
                                              _Float16* __restrict__ thT,
                                              _Float16* __restrict__ phT,
                                              _Float16* __restrict__ gC) {
  int bx = blockIdx.x;
  int b = bx >> 5;
  int n0 = (bx & 31) << 7;           // 128 pixels per block
  int w = threadIdx.x >> 5;          // wave: 3 m-tiles each (24 total)
  int lane = threadIdx.x & 31;
  int rbase = (lane >> 4) << 3;
  int col = lane & 15;
  for (int nt = 0; nt < 8; ++nt) {
    int nb = n0 + nt * 16;
    const _Float16* xrow = xh + ((size_t)b * N + nb) * C;
    v8f acc0 = vzero(), acc1 = vzero(), acc2 = vzero();
#pragma unroll
    for (int kc = 0; kc < 8; ++kc) {
      v16h bf = frag_B(xrow, C, kc * 32);
      v16h a0 = frag_A(Wqkv + (size_t)(w * 3 + 0) * 16 * C, C, kc * 32);
      v16h a1 = frag_A(Wqkv + (size_t)(w * 3 + 1) * 16 * C, C, kc * 32);
      v16h a2 = frag_A(Wqkv + (size_t)(w * 3 + 2) * 16 * C, C, kc * 32);
      acc0 = wmma(a0, bf, acc0);
      acc1 = wmma(a1, bf, acc1);
      acc2 = wmma(a2, bf, acc2);
    }
    int n = nb + col;
#pragma unroll
    for (int i = 0; i < 3; ++i) {
      v8f acc = (i == 0) ? acc0 : (i == 1) ? acc1 : acc2;
      int mbase = (w * 3 + i) * 16 + rbase;
#pragma unroll
      for (int j = 0; j < 8; ++j) {
        int m = mbase + j;
        float v = acc[j] + bqkv[m];
        if (m < 128)
          thT[((size_t)b * N + n) * Ci + m] = (_Float16)v;
        else if (m < 256)
          phT[((size_t)b * N + n) * Ci + (m - 128)] = (_Float16)v;
        else
          gC[((size_t)b * Ci + (m - 256)) * N + n] = (_Float16)v;
      }
    }
  }
}

// ---------------- flash attention: softmax(thT phTᵀ / sqrt(Ci)) @ g -> yT [B,N,Ci] f16 ----------------
// TDM double-buffers the phi [32xCi] and g [Cix32] tiles into LDS while WMMAs run.
__global__ __launch_bounds__(256) void k_attn(const _Float16* __restrict__ thT,
                                              const _Float16* __restrict__ phT,
                                              const _Float16* __restrict__ gC,
                                              _Float16* __restrict__ yT) {
  __shared__ _Float16 ph_s[2][32][128];  // key rows x Ci, ping-pong
  __shared__ _Float16 g_s[2][128][32];   // Ci rows x key cols, ping-pong
  __shared__ _Float16 P_s[8][16][32];    // per-wave P staging (C-layout -> A-layout)
  int bx = blockIdx.x;
  int b = bx >> 5;
  int n0 = (bx & 31) << 7;             // 128 query rows per block
  int t = threadIdx.x;
  int w = t >> 5, lane = t & 31;
  int rbase = (lane >> 4) << 3, col = lane & 15;
  int nw = n0 + w * 16;                // this wave's 16 query rows
  const float scale = 0.0883883476483184f;  // 1/sqrt(128)

  const _Float16* thbase = thT + ((size_t)b * N + nw) * Ci;
  v16h aTh[4];
#pragma unroll
  for (int kc = 0; kc < 4; ++kc) aTh[kc] = frag_A(thbase, Ci, kc * 32);

  v8f O[8];
#pragma unroll
  for (int i = 0; i < 8; ++i) O[i] = vzero();
  float mrow[8], lrow[8];
#pragma unroll
  for (int j = 0; j < 8; ++j) { mrow[j] = -1e30f; lrow[j] = 0.f; }

  const _Float16* phbase = phT + (size_t)b * N * Ci;
  const _Float16* gbase = gC + (size_t)b * Ci * N;

#if HAVE_TDM
  bool issuer = (t < 32);  // wave 0 drives the TDM
  if (issuer) {
    tdm_load_2d(lds_byte_addr(&ph_s[0][0][0]), phbase, grp1_phi());
    tdm_load_2d(lds_byte_addr(&g_s[0][0][0]), gbase, grp1_g());
  }
#else
  int pr = t >> 3, pc = (t & 7) * 16;   // phi tile copy: 32 rows x 128 halves
  int gr = t >> 1, gcc = (t & 1) * 16;  // g tile copy: 128 rows x 32 halves
#endif

  const int NIT = N / 32;
  for (int it = 0; it < NIT; ++it) {
    int buf = it & 1;
    __syncthreads();  // all waves done computing on buf^1
#if HAVE_TDM
    if (issuer) {
      if (it + 1 < NIT) {
        int m1 = (it + 1) * 32;
        tdm_load_2d(lds_byte_addr(&ph_s[buf ^ 1][0][0]), phbase + (size_t)m1 * Ci, grp1_phi());
        tdm_load_2d(lds_byte_addr(&g_s[buf ^ 1][0][0]), gbase + m1, grp1_g());
        wait_tensorcnt2();  // pair for tile `it` complete
      } else {
        wait_tensorcnt0();
      }
    }
#else
    {
      int m0 = it * 32;
      const v8h* s = (const v8h*)(phbase + (size_t)(m0 + pr) * Ci + pc);
      *(v8h*)&ph_s[buf][pr][pc] = s[0];
      *(v8h*)&ph_s[buf][pr][pc + 8] = s[1];
      const v8h* sg = (const v8h*)(gbase + (size_t)gr * N + m0 + gcc);
      *(v8h*)&g_s[buf][gr][gcc] = sg[0];
      *(v8h*)&g_s[buf][gr][gcc + 8] = sg[1];
      if (it + 1 < NIT) {
        __builtin_prefetch(phbase + (size_t)(m0 + 32 + pr) * Ci + pc, 0, 0);
        __builtin_prefetch(gbase + (size_t)gr * N + m0 + 32 + gcc, 0, 0);
      }
    }
#endif
    __syncthreads();  // tile `it` visible in LDS to all waves

    // S tile (16 rows x 32 key cols), K = Ci = 128
    v8f s0 = vzero(), s1 = vzero();
#pragma unroll
    for (int kc = 0; kc < 4; ++kc) {
      v16h b0 = frag_B(&ph_s[buf][0][0], 128, kc * 32);
      v16h b1 = frag_B(&ph_s[buf][16][0], 128, kc * 32);
      s0 = wmma(aTh[kc], b0, s0);
      s1 = wmma(aTh[kc], b1, s1);
    }

    // online softmax update (row reductions within 16-lane half)
    float alpha[8];
#pragma unroll
    for (int j = 0; j < 8; ++j) {
      float a = s0[j] * scale, bb = s1[j] * scale;
      float v = fmaxf(a, bb);
      v = fmaxf(v, __shfl_xor(v, 1));
      v = fmaxf(v, __shfl_xor(v, 2));
      v = fmaxf(v, __shfl_xor(v, 4));
      v = fmaxf(v, __shfl_xor(v, 8));
      float mnew = fmaxf(mrow[j], v);
      alpha[j] = __expf(mrow[j] - mnew);
      float p0 = __expf(a - mnew);
      float p1 = __expf(bb - mnew);
      s0[j] = p0;
      s1[j] = p1;
      float rs = p0 + p1;
      rs += __shfl_xor(rs, 1);
      rs += __shfl_xor(rs, 2);
      rs += __shfl_xor(rs, 4);
      rs += __shfl_xor(rs, 8);
      lrow[j] = lrow[j] * alpha[j] + rs;
      mrow[j] = mnew;
    }
#pragma unroll
    for (int i = 0; i < 8; ++i)
#pragma unroll
      for (int j = 0; j < 8; ++j) O[i][j] *= alpha[j];

    // P (C-layout fp32) -> LDS -> A-layout f16 (same-wave round trip)
#pragma unroll
    for (int j = 0; j < 8; ++j) {
      P_s[w][rbase + j][col] = (_Float16)s0[j];
      P_s[w][rbase + j][16 + col] = (_Float16)s1[j];
    }
    v16h pA = frag_A(&P_s[w][0][0], 32, 0);
#pragma unroll
    for (int ct = 0; ct < 8; ++ct) {
      v16h vb = frag_B(&g_s[buf][ct * 16][0], 32, 0);
      O[ct] = wmma(pA, vb, O[ct]);
    }
  }

  // normalize, store yT[b][n][c] f16
#pragma unroll
  for (int j = 0; j < 8; ++j) lrow[j] = 1.f / lrow[j];
#pragma unroll
  for (int ct = 0; ct < 8; ++ct) {
#pragma unroll
    for (int j = 0; j < 8; ++j) {
      float v = O[ct][j] * lrow[j];
      yT[((size_t)b * N + nw + rbase + j) * Ci + ct * 16 + col] = (_Float16)v;
    }
  }
}

// ---------------- output projection + BN partial stats ----------------
__global__ __launch_bounds__(256) void k_oproj(const _Float16* __restrict__ yT,
                                               const _Float16* __restrict__ Wout,
                                               const float* __restrict__ out_b,
                                               float* __restrict__ yf,
                                               float* __restrict__ ssum,
                                               float* __restrict__ ssq) {
  int bx = blockIdx.x;
  int b = bx >> 5;
  int n0 = (bx & 31) << 7;
  int w = threadIdx.x >> 5, lane = threadIdx.x & 31;
  int rbase = (lane >> 4) << 3, col = lane & 15;
  float psum[2][8], psq[2][8];
#pragma unroll
  for (int i = 0; i < 2; ++i)
#pragma unroll
    for (int j = 0; j < 8; ++j) { psum[i][j] = 0.f; psq[i][j] = 0.f; }

  for (int nt = 0; nt < 8; ++nt) {
    const _Float16* yrow = yT + ((size_t)b * N + n0 + nt * 16) * Ci;
    if (nt + 1 < 8)
      __builtin_prefetch(yrow + 16 * Ci + (size_t)col * Ci, 0, 0);
    v8f acc0 = vzero(), acc1 = vzero();
#pragma unroll
    for (int kc = 0; kc < 4; ++kc) {
      v16h bf = frag_B(yrow, Ci, kc * 32);
      v16h a0 = frag_A(Wout + (size_t)(w * 2 + 0) * 16 * Ci, Ci, kc * 32);
      v16h a1 = frag_A(Wout + (size_t)(w * 2 + 1) * 16 * Ci, Ci, kc * 32);
      acc0 = wmma(a0, bf, acc0);
      acc1 = wmma(a1, bf, acc1);
    }
    int n = n0 + nt * 16 + col;
#pragma unroll
    for (int i = 0; i < 2; ++i) {
      v8f acc = i ? acc1 : acc0;
#pragma unroll
      for (int j = 0; j < 8; ++j) {
        int m = (w * 2 + i) * 16 + rbase + j;
        float v = acc[j] + out_b[m];
        yf[((size_t)b * C + m) * N + n] = v;
        psum[i][j] += v;
        psq[i][j] += v * v;
      }
    }
  }
#pragma unroll
  for (int i = 0; i < 2; ++i)
#pragma unroll
    for (int j = 0; j < 8; ++j) {
      float s = psum[i][j], q = psq[i][j];
      s += __shfl_xor(s, 1); s += __shfl_xor(s, 2); s += __shfl_xor(s, 4); s += __shfl_xor(s, 8);
      q += __shfl_xor(q, 1); q += __shfl_xor(q, 2); q += __shfl_xor(q, 4); q += __shfl_xor(q, 8);
      if (col == 0) {
        int m = (w * 2 + i) * 16 + rbase + j;
        atomicAdd(&ssum[m], s);
        atomicAdd(&ssq[m], q);
      }
    }
}

// ---------------- BN finalize + residual ----------------
__global__ __launch_bounds__(256) void k_final(const float* __restrict__ x,
                                               const float* __restrict__ yf,
                                               const float* __restrict__ ssum,
                                               const float* __restrict__ ssq,
                                               const float* __restrict__ gamma,
                                               const float* __restrict__ beta,
                                               float* __restrict__ out) {
  const float inv = 1.0f / (float)(Bb * N);
  size_t total = (size_t)Bb * C * N;
  for (size_t i = (size_t)blockIdx.x * blockDim.x + threadIdx.x; i < total;
       i += (size_t)gridDim.x * blockDim.x) {
    int c = (int)((i >> 12) & 255);
    float mean = ssum[c] * inv;
    float var = ssq[c] * inv - mean * mean;
    float r = rsqrtf(var + 1e-5f);
    out[i] = x[i] + (yf[i] - mean) * r * gamma[c] + beta[c];
  }
}

extern "C" void kernel_launch(void* const* d_in, const int* in_sizes, int n_in,
                              void* d_out, int out_size, void* d_ws, size_t ws_size,
                              hipStream_t stream) {
  (void)in_sizes; (void)n_in; (void)out_size; (void)ws_size;
  const float* x   = (const float*)d_in[0];
  const float* tw  = (const float*)d_in[1];
  const float* tb  = (const float*)d_in[2];
  const float* pw  = (const float*)d_in[3];
  const float* pb  = (const float*)d_in[4];
  const float* gw  = (const float*)d_in[5];
  const float* gb  = (const float*)d_in[6];
  const float* ow  = (const float*)d_in[7];
  const float* ob  = (const float*)d_in[8];
  const float* gma = (const float*)d_in[9];
  const float* bta = (const float*)d_in[10];

  char* ws = (char*)d_ws;
  size_t off = 0;
  auto alloc = [&](size_t bytes) -> void* {
    void* p = ws + off;
    off += (bytes + 255) & ~(size_t)255;
    return p;
  };
  _Float16* xh   = (_Float16*)alloc((size_t)Bb * N * C * 2);   // 16 MB
  _Float16* Wqkv = (_Float16*)alloc((size_t)384 * C * 2);
  _Float16* Wout = (_Float16*)alloc((size_t)C * Ci * 2);
  float*    bqkv = (float*)alloc((size_t)384 * 4);
  _Float16* thT  = (_Float16*)alloc((size_t)Bb * N * Ci * 2);  // 8 MB
  _Float16* phT  = (_Float16*)alloc((size_t)Bb * N * Ci * 2);  // 8 MB
  _Float16* gCb  = (_Float16*)alloc((size_t)Bb * Ci * N * 2);  // 8 MB
  _Float16* yT   = (_Float16*)alloc((size_t)Bb * N * Ci * 2);  // 8 MB
  float*    yf   = (float*)alloc((size_t)Bb * C * N * 4);      // 33.5 MB
  float*    stat = (float*)alloc((size_t)512 * 4);
  float* ssum = stat;
  float* ssq  = stat + 256;

  k_zero<<<1, 512, 0, stream>>>(stat);
  int nprep = 384 * C + C * Ci + 384;
  k_prep_w<<<(nprep + 255) / 256, 256, 0, stream>>>(tw, tb, pw, pb, gw, gb, ow, Wqkv, Wout, bqkv);
  k_prep_x<<<Bb * 8 * 64, 256, 0, stream>>>(x, xh);
  k_proj<<<Bb * 32, 256, 0, stream>>>(xh, Wqkv, bqkv, thT, phT, gCb);
  k_attn<<<Bb * 32, 256, 0, stream>>>(thT, phT, gCb, yT);
  k_oproj<<<Bb * 32, 256, 0, stream>>>(yT, Wout, ob, yf, ssum, ssq);
  k_final<<<8192, 256, 0, stream>>>(x, yf, ssum, ssq, gma, bta, (float*)d_out);
}
